// MyLSTM_28406913696410
// MI455X (gfx1250) — matrix operand order repaired and checked by reading, not compiled, not run on hardware
//
#include <hip/hip_runtime.h>
#include <hip/hip_bf16.h>

typedef __attribute__((ext_vector_type(16))) _Float16 v16h;
typedef __attribute__((ext_vector_type(8)))  _Float16 v8h;
typedef __attribute__((ext_vector_type(8)))  float    v8f;

#define Bb   256
#define Tt   256
#define Ll   256
#define FourL 1024
#define BT   (Bb*Tt)

// ---------------------------------------------------------------------------
// Pack a (K x 1024) f32 weight matrix into WMMA B-fragment layout, f16.
// Fragment (kt,j): 32 lanes x 16 f16.  B element e: k = kt*32 + (lane>=16)*8 + (e&7) + (e>>3)*16,
// n = j*16 + (lane&15).  One wave per fragment.
// ---------------------------------------------------------------------------
__global__ void pack_bfrag_kernel(const float* __restrict__ src,
                                  _Float16* __restrict__ dst, int K) {
  int wave = threadIdx.x >> 5, lane = threadIdx.x & 31;
  int frag = blockIdx.x * 8 + wave;
  int KT = K >> 5;
  int kt = frag >> 6, j = frag & 63;
  if (kt >= KT) return;
  int nl = lane & 15, hi = lane >> 4;
  int n = j * 16 + nl;
  _Float16* d = dst + ((size_t)frag * 32 + lane) * 16;
#pragma unroll
  for (int e = 0; e < 16; ++e) {
    int k = kt * 32 + hi * 8 + (e & 7) + ((e >> 3) << 4);
    d[e] = (_Float16)src[(size_t)k * FourL + n];
  }
}

// ---------------------------------------------------------------------------
// Keras Masking(mask_value=-1): maskT[t*B + b] = any(x[b,t,:] != -1)
// (stored t-major so one step's 16 rows are a contiguous 64B line)
// ---------------------------------------------------------------------------
__global__ void mask_kernel(const float* __restrict__ x,
                            float* __restrict__ maskT, int F) {
  int idx = blockIdx.x * 256 + threadIdx.x;   // b*T + t
  int b = idx >> 8, t = idx & 255;
  const float* p = x + (size_t)idx * F;
  float any = 0.f;
  for (int f = 0; f < F; ++f) {
    if (p[f] != -1.0f) { any = 1.f; break; }
  }
  maskT[t * Bb + b] = any;
}

// ---------------------------------------------------------------------------
// xproj GEMM: (BT x F) @ (F x 1024) + bias.  Output stored f16 in the exact
// WMMA C-fragment layout the recurrent kernel consumes:
//   xp[(((t*16 + btile)*64 + j)*32 + lane)*8 + r]
// so the LSTM reads one b128 per lane per tile.  The (b,t) transpose is
// absorbed here as 8 scattered b16 stores per lane (parallel pass).
// grid = (BT/16, 64/8), block = 256 (8 waves, one 16x16 tile each).
// ---------------------------------------------------------------------------
__global__ void xproj_gemm_kernel(const float* __restrict__ x,
                                  const _Float16* __restrict__ wfrag,
                                  const float* __restrict__ bias,
                                  _Float16* __restrict__ xp, int F) {
  int wave = threadIdx.x >> 5, lane = threadIdx.x & 31;
  int rowtile = blockIdx.x;                 // rows = rowtile*16..+15 (same b)
  int j = blockIdx.y * 8 + wave;            // output column tile 0..63
  int b  = rowtile >> 4;
  int t0 = (rowtile & 15) * 16;
  int nl = lane & 15, hi = lane >> 4;
  int col = j * 16 + nl;
  float bv = bias[col];
  int KT = F >> 5;
  v8f acc = {};
  for (int kt = 0; kt < KT; ++kt) {
    // A fragment: m = lane&15, k = kt*32 + hi*8 + (e&7) + (e>>3)*16
    const float* xr = x + (size_t)(rowtile * 16 + nl) * F + kt * 32 + hi * 8;
    v16h a;
#pragma unroll
    for (int e = 0; e < 16; ++e) a[e] = (_Float16)xr[(e & 7) + ((e >> 3) << 4)];
    v16h bf = *(const v16h*)(wfrag + (((size_t)kt * 64 + j) * 32 + lane) * 16);
    acc = __builtin_amdgcn_wmma_f32_16x16x32_f16(false, a, false, bf,
                                                 (short)0, acc, false, false);
  }
  // scatter into consumer fragment layout
  int bt = b >> 4;
  int mb = b & 15;                          // this b's row inside LSTM tile
  int lanep = (mb >> 3) * 16 + nl;
  int rp = mb & 7;
#pragma unroll
  for (int r = 0; r < 8; ++r) {
    int t = t0 + r + hi * 8;                // GEMM row m -> time index
    xp[((((size_t)t * 16 + bt) * 64 + j) * 32 + lanep) * 8 + rp] =
        (_Float16)(acc[r] + bv);
  }
}

// ---------------------------------------------------------------------------
// Recurrent LSTM kernel. 32 workgroups: blocks 0..15 -> LSTM1 batch tiles,
// 16..31 -> LSTM2. Each WG owns 16 batch rows and all 1024 gate columns ->
// no inter-WG traffic. h lives in LDS in A-fragment layout; c state lives in
// accumulator VGPRs. 8 waves, each: 2 hidden chunks x 4 gates = 64 WMMA/step.
// xp fragments (b128/lane) are issued before the WMMA K-loop to hide latency.
// ---------------------------------------------------------------------------
__global__ void __launch_bounds__(256)
lstm_kernel(const _Float16* __restrict__ xp1, const _Float16* __restrict__ U1f,
            const float* __restrict__ mask1,
            const _Float16* __restrict__ xp2, const _Float16* __restrict__ U2f,
            const float* __restrict__ mask2,
            float* __restrict__ merged) {
  __shared__ _Float16 hbuf[8 * 32 * 16];   // h (16 x 256) in A-fragment layout

  int sel = blockIdx.x >> 4;
  int bt  = blockIdx.x & 15;
  const _Float16* xp    = sel ? xp2   : xp1;
  const _Float16* Uf    = sel ? U2f   : U1f;
  const float*    maskT = sel ? mask2 : mask1;
  int coff = sel ? Ll : 0;

  int wave = threadIdx.x >> 5, lane = threadIdx.x & 31;
  int nl = lane & 15, hi = lane >> 4;
  int b0 = bt * 16;
  int q0 = wave * 2;                       // this wave's hidden chunks: q0,q0+1

  for (int i = threadIdx.x; i < 8 * 32 * 16; i += 256) hbuf[i] = (_Float16)0.f;
  __syncthreads();

  v8f cst[2] = {};                         // cell state, f32, VGPR-resident
  float hs[2][8];                          // staged h_new across the barrier

  for (int t = 0; t < Tt; ++t) {
    // ---- issue xp fragment loads first: latency hidden under WMMA K-loop
    v8h xpf[8];
    size_t fbase = ((size_t)t * 16 + bt) * 64;
#pragma unroll
    for (int tt = 0; tt < 8; ++tt) {
      int j = (tt & 3) * 16 + q0 + (tt >> 2);
      xpf[tt] = *(const v8h*)(xp + ((fbase + j) * 32 + lane) * 8);
    }
    // per-step row masks (one 64B line, broadcast across WG)
    float mv[8];
#pragma unroll
    for (int r = 0; r < 8; ++r) mv[r] = maskT[t * Bb + b0 + r + hi * 8];

    // ---- z = h @ U
    v8f acc[8] = {};
    for (int kt = 0; kt < 8; ++kt) {
      v16h a = *(const v16h*)(&hbuf[(kt * 32 + lane) * 16]);
      if (kt < 7) {
        __builtin_prefetch((const void*)(Uf + (((size_t)(kt + 1) * 64 + q0) * 32 + lane) * 16), 0, 1);
      }
#pragma unroll
      for (int tt = 0; tt < 8; ++tt) {
        int j = (tt & 3) * 16 + q0 + (tt >> 2);
        v16h bf = *(const v16h*)(Uf + (((size_t)kt * 64 + j) * 32 + lane) * 16);
        acc[tt] = __builtin_amdgcn_wmma_f32_16x16x32_f16(false, a, false, bf,
                                                         (short)0, acc[tt],
                                                         false, false);
      }
    }

    // ---- gates + state update (reads hbuf for carry-through: before barrier)
#pragma unroll
    for (int c = 0; c < 2; ++c) {
      int q  = q0 + c;
      int k  = q * 16 + nl;                // hidden column 0..255
      int kt = k >> 5;
      int k5 = k & 31;
      int e   = (k5 & 7) + ((k5 >> 4) << 3);
      int klo = (k5 & 8) ? 16 : 0;
#pragma unroll
      for (int r = 0; r < 8; ++r) {
        int m = r + hi * 8;
        int brow = b0 + m;
        float zi = acc[c * 4 + 0][r] + (float)xpf[c * 4 + 0][r];
        float zf = acc[c * 4 + 1][r] + (float)xpf[c * 4 + 1][r];
        float zg = acc[c * 4 + 2][r] + (float)xpf[c * 4 + 2][r];
        float zo = acc[c * 4 + 3][r] + (float)xpf[c * 4 + 3][r];
        float si = 1.f / (1.f + __expf(-zi));
        float sf = 1.f / (1.f + __expf(-zf));
        float so = 1.f / (1.f + __expf(-zo));
        float rg = zg > 0.f ? zg : 0.f;
        float cn = sf * cst[c][r] + si * rg;
        float hn = so * (cn > 0.f ? cn : 0.f);
        float hold = (float)hbuf[(kt * 32 + m + klo) * 16 + e];
        float cnew = (mv[r] != 0.f) ? cn : cst[c][r];
        float hnew = (mv[r] != 0.f) ? hn : hold;
        cst[c][r] = cnew;
        hs[c][r]  = hnew;
        merged[((size_t)brow * Tt + t) * (2 * Ll) + coff + k] = hnew;
      }
    }
    __syncthreads();                       // all A-frag / h_old reads done
#pragma unroll
    for (int c = 0; c < 2; ++c) {
      int q = q0 + c;
      int k = q * 16 + nl;
      int kt = k >> 5;
      int k5 = k & 31;
      int e   = (k5 & 7) + ((k5 >> 4) << 3);
      int klo = (k5 & 8) ? 16 : 0;
#pragma unroll
      for (int r = 0; r < 8; ++r) {
        int m = r + hi * 8;
        hbuf[(kt * 32 + m + klo) * 16 + e] = (_Float16)hs[c][r];
      }
    }
    __syncthreads();                       // h_t visible for step t+1
  }
}

// ---------------------------------------------------------------------------
// Attention + output head: one block per batch row.
// ---------------------------------------------------------------------------
__global__ void attn_kernel(const float* __restrict__ merged,
                            const float* __restrict__ d1w, const float* __restrict__ d1b,
                            const float* __restrict__ d2w, const float* __restrict__ d2b,
                            float* __restrict__ out) {
  __shared__ float sm[256];
  __shared__ float red[256];
  int b = blockIdx.x, tid = threadIdx.x;
  const float* mb = merged + (size_t)b * Tt * (2 * Ll);

  // score_t = tanh(merged[b,t,:] . d1w + d1b)   (float4 loads)
  const float4* row = (const float4*)(mb + (size_t)tid * (2 * Ll));
  const float4* w4  = (const float4*)d1w;
  float s = 0.f;
  for (int d = 0; d < (2 * Ll) / 4; ++d) {
    float4 rv = row[d], wv = w4[d];
    s += rv.x * wv.x + rv.y * wv.y + rv.z * wv.z + rv.w * wv.w;
  }
  s = tanhf(s + d1b[0]);
  red[tid] = s;
  __syncthreads();
  for (int off = 128; off > 0; off >>= 1) {
    if (tid < off) red[tid] = fmaxf(red[tid], red[tid + off]);
    __syncthreads();
  }
  float mx = red[0];
  __syncthreads();
  float e = __expf(s - mx);
  sm[tid]  = e;
  red[tid] = e;
  __syncthreads();
  for (int off = 128; off > 0; off >>= 1) {
    if (tid < off) red[tid] += red[tid + off];
    __syncthreads();
  }
  float inv = 1.f / red[0];
  __syncthreads();

  // rep[d] = sum_t merged[b,t,d]*attn[t]; thread owns d=tid and d=tid+256
  float r0 = 0.f, r1 = 0.f;
  for (int t = 0; t < Tt; ++t) {
    float a = sm[t] * inv;
    r0 += mb[(size_t)t * (2 * Ll) + tid      ] * a;
    r1 += mb[(size_t)t * (2 * Ll) + tid + 256] * a;
  }
  float p = r0 * d2w[tid] + r1 * d2w[tid + 256];
  red[tid] = p;
  __syncthreads();
  for (int off = 128; off > 0; off >>= 1) {
    if (tid < off) red[tid] += red[tid + off];
    __syncthreads();
  }
  if (tid == 0) out[b] = 1.f / (1.f + __expf(-(red[0] + d2b[0])));
}

// ---------------------------------------------------------------------------
extern "C" void kernel_launch(void* const* d_in, const int* in_sizes, int n_in,
                              void* d_out, int out_size, void* d_ws, size_t ws_size,
                              hipStream_t stream) {
  (void)in_sizes; (void)n_in; (void)out_size; (void)ws_size;
  const float* inputs1 = (const float*)d_in[0];
  const float* inputs2 = (const float*)d_in[1];
  const float* W1  = (const float*)d_in[2];
  const float* U1  = (const float*)d_in[3];
  const float* b1  = (const float*)d_in[4];
  const float* W2  = (const float*)d_in[5];
  const float* U2  = (const float*)d_in[6];
  const float* b2  = (const float*)d_in[7];
  const float* d1w = (const float*)d_in[8];
  const float* d1b = (const float*)d_in[9];
  const float* d2w = (const float*)d_in[10];
  const float* d2b = (const float*)d_in[11];
  float* out = (float*)d_out;

  // workspace layout (bytes, 256-aligned)
  char* ws = (char*)d_ws;
  size_t o = 0;
  auto take = [&](size_t bytes) { char* p = ws + o; o = (o + bytes + 255) & ~(size_t)255; return p; };
  _Float16* W1f   = (_Float16*)take((size_t)4 * 64 * 32 * 16 * 2);     // 256 KB
  _Float16* W2f   = (_Float16*)take((size_t)2 * 64 * 32 * 16 * 2);     // 128 KB
  _Float16* U1f   = (_Float16*)take((size_t)8 * 64 * 32 * 16 * 2);     // 512 KB
  _Float16* U2f   = (_Float16*)take((size_t)8 * 64 * 32 * 16 * 2);     // 512 KB
  float*    mask1 = (float*)take((size_t)BT * 4);                      // 256 KB
  float*    mask2 = (float*)take((size_t)BT * 4);                      // 256 KB
  _Float16* xp1   = (_Float16*)take((size_t)BT * FourL * 2);           // 128 MB
  _Float16* xp2   = (_Float16*)take((size_t)BT * FourL * 2);           // 128 MB
  float*    mrg   = (float*)take((size_t)BT * 2 * Ll * 4);             // 128 MB

  pack_bfrag_kernel<<<(4 * 64) / 8, 256, 0, stream>>>(W1, W1f, 128);
  pack_bfrag_kernel<<<(2 * 64) / 8, 256, 0, stream>>>(W2, W2f, 64);
  pack_bfrag_kernel<<<(8 * 64) / 8, 256, 0, stream>>>(U1, U1f, 256);
  pack_bfrag_kernel<<<(8 * 64) / 8, 256, 0, stream>>>(U2, U2f, 256);

  mask_kernel<<<BT / 256, 256, 0, stream>>>(inputs1, mask1, 128);
  mask_kernel<<<BT / 256, 256, 0, stream>>>(inputs2, mask2, 64);

  xproj_gemm_kernel<<<dim3(BT / 16, 8), 256, 0, stream>>>(inputs1, W1f, b1, xp1, 128);
  xproj_gemm_kernel<<<dim3(BT / 16, 8), 256, 0, stream>>>(inputs2, W2f, b2, xp2, 64);

  lstm_kernel<<<32, 256, 0, stream>>>(xp1, U1f, mask1, xp2, U2f, mask2, mrg);

  attn_kernel<<<Bb, 256, 0, stream>>>(mrg, d1w, d1b, d2w, d2b, out);
}
